// MelToWaveform_77670188580906
// MI455X (gfx1250) — matrix operand order repaired and checked by reading, not compiled, not run on hardware
//
#include <hip/hip_runtime.h>
#include <math.h>

// ---------------- problem constants ----------------
#define N_FFT   1024
#define N_MELS  80
#define HOP     256
#define N_ITER  32
#define N_FREQ  513
#define PADW    512
#define NB      8
#define NT      1024                       // frames
#define L_FULL  (N_FFT + HOP * (NT - 1))   // 262912
#define L_CROP  (L_FULL - N_FFT)           // 261888
#define MOM     (0.99f / 1.99f)
#define KPAD    96                         // 80 -> 3 x 32
#define FPAD    528                        // 513 -> 33 x 16

typedef _Float16 v16h __attribute__((ext_vector_type(16)));
typedef _Float16 h8   __attribute__((ext_vector_type(8)));
typedef float    v8f  __attribute__((ext_vector_type(8)));

// ---------------- helpers ----------------
__device__ __forceinline__ float hannw(int j) {
    return 0.5f - 0.5f * __cosf(6.2831853071795864f * (float)j * (1.0f / 1024.0f));
}

__device__ __forceinline__ float u01hash(unsigned x) {
    x ^= x >> 16; x *= 0x7feb352dU;
    x ^= x >> 15; x *= 0x846ca68bU;
    x ^= x >> 16;
    return (float)(x >> 8) * (1.0f / 16777216.0f);
}

// Shared-memory radix-2 DIT FFT, 1024 points, 256 threads, bit-reversed input,
// natural-order output. tw[k] = exp(-2*pi*i*k/1024); inverse uses conjugate.
__device__ __forceinline__ void fft1024(float2* a, const float2* tw, int tid, bool inverse) {
    for (int s = 1; s <= 10; ++s) {
        const int half = 1 << (s - 1);
        __syncthreads();
        #pragma unroll 1
        for (int idx = tid; idx < 512; idx += 256) {
            const int pos = idx & (half - 1);
            const int grp = idx >> (s - 1);
            const int i0 = (grp << s) + pos;
            const int i1 = i0 + half;
            float2 w = tw[pos << (10 - s)];
            const float wim = inverse ? -w.y : w.y;
            const float2 x = a[i1];
            const float tre = w.x * x.x - wim * x.y;
            const float tim = w.x * x.y + wim * x.x;
            const float2 u = a[i0];
            a[i0] = make_float2(u.x + tre, u.y + tim);
            a[i1] = make_float2(u.x - tre, u.y - tim);
        }
    }
    __syncthreads();
}

__device__ __forceinline__ void load_twiddles(float2* tw, int tid) {
    for (int k = tid; k < 512; k += 256) {
        float sn, cs;
        __sincosf(-6.2831853071795864f * (float)k * (1.0f / 1024.0f), &sn, &cs);
        tw[k] = make_float2(cs, sn);
    }
}

// ---------------- pinv(fb.T) = fb @ inv(fb^T fb) ----------------
__global__ void m2w_gram(const float* __restrict__ fb, float* __restrict__ G) {
    for (int p = threadIdx.x; p < N_MELS * N_MELS; p += 256) {
        const int i = p / N_MELS, j = p % N_MELS;
        float acc = 0.0f;
        for (int f = 0; f < N_FREQ; ++f)
            acc += fb[f * N_MELS + i] * fb[f * N_MELS + j];
        G[p] = acc;
    }
}

__global__ void m2w_ginv(const float* __restrict__ G, float* __restrict__ Gi) {
    __shared__ float M[N_MELS][161];  // 80 x 160 augmented (pad col to break conflicts)
    __shared__ float fcol[N_MELS];
    const int tid = threadIdx.x;
    for (int p = tid; p < N_MELS * 160; p += 256) {
        const int i = p / 160, j = p % 160;
        M[i][j] = (j < N_MELS) ? G[i * N_MELS + j] : ((j - N_MELS) == i ? 1.0f : 0.0f);
    }
    __syncthreads();
    for (int k = 0; k < N_MELS; ++k) {
        const float rpiv = 1.0f / M[k][k];   // SPD gram: diagonal pivots are safe
        __syncthreads();
        for (int j = tid; j < 160; j += 256) M[k][j] *= rpiv;
        __syncthreads();
        for (int i = tid; i < N_MELS; i += 256) fcol[i] = M[i][k];
        __syncthreads();
        for (int p = tid; p < N_MELS * 160; p += 256) {
            const int i = p / 160, j = p % 160;
            if (i != k) M[i][j] -= fcol[i] * M[k][j];
        }
        __syncthreads();
    }
    for (int p = tid; p < N_MELS * N_MELS; p += 256)
        Gi[p] = M[p / N_MELS][N_MELS + (p % N_MELS)];
}

// P = fb @ Ginv, written directly as zero-padded f16 (FPAD x KPAD) for WMMA-A.
__global__ void m2w_pinv_f16(const float* __restrict__ fb, const float* __restrict__ Gi,
                             _Float16* __restrict__ Ph) {
    const int p = blockIdx.x * 256 + threadIdx.x;
    if (p >= FPAD * KPAD) return;
    const int f = p / KPAD, m = p % KPAD;
    float acc = 0.0f;
    if (f < N_FREQ && m < N_MELS)
        for (int k = 0; k < N_MELS; ++k)
            acc += fb[f * N_MELS + k] * Gi[k * N_MELS + m];
    Ph[p] = (_Float16)acc;
}

// mel (b,m,t) f32 -> melTh (b,t,kpad) f16, K contiguous + zero padded (WMMA-B staging)
__global__ void m2w_melT_f16(const float* __restrict__ mel, _Float16* __restrict__ melTh) {
    const int p = blockIdx.x * 256 + threadIdx.x;
    if (p >= NB * NT * KPAD) return;
    const int k  = p % KPAD;
    const int bt = p / KPAD;
    const int t  = bt & (NT - 1);
    const int b  = bt >> 10;
    melTh[p] = (k < N_MELS) ? (_Float16)mel[(b * N_MELS + k) * NT + t] : (_Float16)0.0f;
}

// ---------------- mag = sqrt(relu(P @ mel)) via WMMA ----------------
// 256-thread block = 8 waves; each wave owns one 16(f) x 64(t) strip: one shared
// A fragment per k-step feeds 4 accumulators. All fragment loads are unconditional
// vector loads from the padded f16 staging buffers.
// 16-bit A (16x32) layout used here (not the ISA table order): lane l holds row M = l&15; element e holds
// K = k0+16*(l>>4)+e (matching B)  -> two h8 loads.
// B (32x16): lane n holds col N = n&15; element e holds K = k0+16*(n>>4)+e -> one v16h load.
__global__ void __launch_bounds__(256)
m2w_mag_wmma(const _Float16* __restrict__ Ph, const _Float16* __restrict__ melTh,
             float* __restrict__ mag) {
    const int wave = threadIdx.x >> 5;
    const int lane = threadIdx.x & 31;
    const int blk  = blockIdx.x;          // NB * 33 * 2 blocks
    const int tb   = blk & 1;             // t half (512 each)
    const int ft   = (blk >> 1) % 33;
    const int b    = blk / (33 * 2);
    const int low  = lane & 15;
    const int hi   = lane >> 4;
    const int f0   = ft * 16;
    const int tbase = tb * 512 + wave * 64;   // 4 t-tiles of 16 from here

    const _Float16* arow = Ph + (f0 + low) * KPAD;
    v8f c0 = {}, c1 = {}, c2 = {}, c3 = {};

    #pragma unroll
    for (int k0 = 0; k0 < KPAD; k0 += 32) {
        union { v16h v; h8 h[2]; } A;
        A.h[0] = *(const h8*)(arow + k0 + 16 * hi);
        A.h[1] = *(const h8*)(arow + k0 + 16 * hi + 8);
        const int kb = k0 + 16 * hi;
        const v16h B0 = *(const v16h*)(melTh + ((size_t)(b * NT) + tbase +  0 + low) * KPAD + kb);
        const v16h B1 = *(const v16h*)(melTh + ((size_t)(b * NT) + tbase + 16 + low) * KPAD + kb);
        const v16h B2 = *(const v16h*)(melTh + ((size_t)(b * NT) + tbase + 32 + low) * KPAD + kb);
        const v16h B3 = *(const v16h*)(melTh + ((size_t)(b * NT) + tbase + 48 + low) * KPAD + kb);
        c0 = __builtin_amdgcn_wmma_f32_16x16x32_f16(false, A.v, false, B0, (short)0, c0, false, false);
        c1 = __builtin_amdgcn_wmma_f32_16x16x32_f16(false, A.v, false, B1, (short)0, c1, false, false);
        c2 = __builtin_amdgcn_wmma_f32_16x16x32_f16(false, A.v, false, B2, (short)0, c2, false, false);
        c3 = __builtin_amdgcn_wmma_f32_16x16x32_f16(false, A.v, false, B3, (short)0, c3, false, false);
    }

    // C/D layout: VGPR i -> M = f0 + 8*hi + i, N = t0 + (lane&15)
    const int f_lo = f0 + 8 * hi;
    #pragma unroll
    for (int q = 0; q < 4; ++q) {
        const v8f* cq = (q == 0) ? &c0 : (q == 1) ? &c1 : (q == 2) ? &c2 : &c3;
        const int t = tbase + q * 16 + low;
        #pragma unroll
        for (int i = 0; i < 8; ++i) {
            const int f = f_lo + i;
            if (f < N_FREQ) {
                const float s = (*cq)[i];
                mag[(b * N_FREQ + f) * NT + t] = (s > 0.0f) ? sqrtf(s) : 0.0f;
            }
        }
    }
}

// ---------------- init angles / tprev ----------------
__global__ void m2w_init(float2* __restrict__ ang, float2* __restrict__ tprev, int n) {
    const int i = blockIdx.x * 256 + threadIdx.x;
    if (i >= n) return;
    ang[i]   = make_float2(u01hash(2u * (unsigned)i + 1u), u01hash(2u * (unsigned)i + 2u));
    tprev[i] = make_float2(0.0f, 0.0f);
}

// ---------------- window-square normalization ----------------
__global__ void m2w_wsq(float* __restrict__ wsq) {
    const int l = blockIdx.x * 256 + threadIdx.x;
    if (l >= L_FULL) return;
    float acc = 0.0f;
    const int thi = l >> 8;
    for (int dt = 0; dt < 4; ++dt) {
        const int t = thi - dt;
        if (t < 0 || t >= NT) continue;
        const int j = l - t * HOP;
        const float w = hannw(j);
        acc += w * w;
    }
    wsq[l] = acc;
}

// ---------------- ISTFT: frames = irfft(mag*ang) * win ----------------
__global__ void m2w_istft(const float* __restrict__ mag, const float2* __restrict__ ang,
                          float* __restrict__ frames) {
    const int bt = blockIdx.x;
    const int t = bt & (NT - 1);
    const int b = bt >> 10;
    const int tid = threadIdx.x;
    __shared__ float2 a[1024];
    __shared__ float2 tw[512];
    load_twiddles(tw, tid);
    for (int j = tid; j < 1024; j += 256) {
        const int f = (j <= 512) ? j : (1024 - j);
        const int idx = (b * N_FREQ + f) * NT + t;
        const float m = mag[idx];
        const float2 an = ang[idx];
        float re = m * an.x, im = m * an.y;
        if (j > 512) im = -im;
        a[__brev((unsigned)j) >> 22] = make_float2(re, im);
    }
    fft1024(a, tw, tid, true);
    const float inv_n = 1.0f / 1024.0f;
    for (int j = tid; j < 1024; j += 256)
        frames[(size_t)bt * N_FFT + j] = a[j].x * inv_n * hannw(j);
}

// ---------------- overlap-add + wsq normalize (gather, no atomics) ----------------
__global__ void m2w_ola(const float* __restrict__ frames, const float* __restrict__ wsq,
                        float* __restrict__ y) {
    const int i = blockIdx.x * 256 + threadIdx.x;
    if (i >= NB * L_FULL) return;
    const int b = i / L_FULL;
    const int l = i - b * L_FULL;
    float acc = 0.0f;
    const int thi = l >> 8;
    #pragma unroll
    for (int dt = 0; dt < 4; ++dt) {
        const int t = thi - dt;
        if (t >= 0 && t < NT) {
            const int j = l - t * HOP;       // (l&255) + 256*dt, always < 1024
            acc += frames[(size_t)(b * NT + t) * N_FFT + j];
        }
    }
    const float w = wsq[l];
    y[i] = acc / ((w > 1e-11f) ? w : 1.0f);
}

// ---------------- STFT + momentum phase update ----------------
__global__ void m2w_stft_update(const float* __restrict__ y, float2* __restrict__ ang,
                                float2* __restrict__ tprev) {
    const int bt = blockIdx.x;
    const int t = bt & (NT - 1);
    const int b = bt >> 10;
    const int tid = threadIdx.x;
    __shared__ float2 a[1024];
    __shared__ float2 tw[512];
    load_twiddles(tw, tid);
    for (int j = tid; j < 1024; j += 256) {
        int p = t * HOP + j - PADW;          // reflect-pad on the cropped signal
        if (p < 0) p = -p;
        else if (p >= L_CROP) p = 2 * L_CROP - 2 - p;
        const float x = y[b * L_FULL + PADW + p];
        a[__brev((unsigned)j) >> 22] = make_float2(x * hannw(j), 0.0f);
    }
    fft1024(a, tw, tid, false);
    for (int f = tid; f < N_FREQ; f += 256) {
        const float2 R = a[f];
        const int idx = (b * N_FREQ + f) * NT + t;
        const float2 tp = tprev[idx];
        const float nre = R.x - MOM * tp.x;
        const float nim = R.y - MOM * tp.y;
        const float inv = 1.0f / (sqrtf(nre * nre + nim * nim) + 1e-16f);
        ang[idx]   = make_float2(nre * inv, nim * inv);
        tprev[idx] = R;
    }
}

// ---------------- final crop ----------------
__global__ void m2w_crop(const float* __restrict__ y, float* __restrict__ out) {
    const int i = blockIdx.x * 256 + threadIdx.x;
    if (i >= NB * L_CROP) return;
    const int b = i / L_CROP;
    const int l = i - b * L_CROP;
    out[i] = y[b * L_FULL + PADW + l];
}

// ---------------- launcher ----------------
extern "C" void kernel_launch(void* const* d_in, const int* in_sizes, int n_in,
                              void* d_out, int out_size, void* d_ws, size_t ws_size,
                              hipStream_t stream) {
    (void)in_sizes; (void)n_in; (void)out_size; (void)ws_size;
    const float* mel = (const float*)d_in[0];   // (8, 80, 1024) f32
    const float* fb  = (const float*)d_in[1];   // (513, 80) f32
    float* out = (float*)d_out;                 // (8, 261888) f32

    char* ws = (char*)d_ws;
    auto alloc = [&](size_t bytes) -> char* {
        char* p = ws;
        ws += (bytes + 255) & ~(size_t)255;
        return p;
    };
    float*     G      = (float*)    alloc((size_t)N_MELS * N_MELS * 4);
    float*     Gi     = (float*)    alloc((size_t)N_MELS * N_MELS * 4);
    _Float16*  Ph     = (_Float16*) alloc((size_t)FPAD * KPAD * 2);
    _Float16*  melTh  = (_Float16*) alloc((size_t)NB * NT * KPAD * 2);
    float*     mag    = (float*)    alloc((size_t)NB * N_FREQ * NT * 4);
    float2*    ang    = (float2*)   alloc((size_t)NB * N_FREQ * NT * 8);
    float2*    tprev  = (float2*)   alloc((size_t)NB * N_FREQ * NT * 8);
    float*     frames = (float*)    alloc((size_t)NB * NT * N_FFT * 4);
    float*     y      = (float*)    alloc((size_t)NB * L_FULL * 4);
    float*     wsq    = (float*)    alloc((size_t)L_FULL * 4);

    const int nspec = NB * N_FREQ * NT;

    m2w_gram<<<1, 256, 0, stream>>>(fb, G);
    m2w_ginv<<<1, 256, 0, stream>>>(G, Gi);
    m2w_pinv_f16<<<(FPAD * KPAD + 255) / 256, 256, 0, stream>>>(fb, Gi, Ph);
    m2w_melT_f16<<<(NB * NT * KPAD + 255) / 256, 256, 0, stream>>>(mel, melTh);
    m2w_mag_wmma<<<NB * 33 * 2, 256, 0, stream>>>(Ph, melTh, mag);
    m2w_init<<<(nspec + 255) / 256, 256, 0, stream>>>(ang, tprev, nspec);
    m2w_wsq<<<(L_FULL + 255) / 256, 256, 0, stream>>>(wsq);

    for (int it = 0; it < N_ITER; ++it) {
        m2w_istft<<<NB * NT, 256, 0, stream>>>(mag, ang, frames);
        m2w_ola<<<(NB * L_FULL + 255) / 256, 256, 0, stream>>>(frames, wsq, y);
        m2w_stft_update<<<NB * NT, 256, 0, stream>>>(y, ang, tprev);
    }
    m2w_istft<<<NB * NT, 256, 0, stream>>>(mag, ang, frames);
    m2w_ola<<<(NB * L_FULL + 255) / 256, 256, 0, stream>>>(frames, wsq, y);
    m2w_crop<<<(NB * L_CROP + 255) / 256, 256, 0, stream>>>(y, out);
}
